// CodeBook_4724464026120
// MI455X (gfx1250) — compile-verified, matrix operand order checked by reading, not run on hardware
//
#include <hip/hip_runtime.h>
#include <hip/hip_bf16.h>
#include <stdint.h>

// ---------------------------------------------------------------------------
// VQ codebook lookup for MI455X (gfx1250, wave32, WMMA).
//   distances[n,k] = ||x_n||^2 + ||e_k||^2 - 2 x_n.e_k ; argmin_k ; gather ;
//   losses. GEMM is compute bound -> bf16 WMMA (fp32 accumulate).
//   v2: 4x2 wave tiling (32x64 per wave), fragment preload for back-to-back
//       WMMA issue, double-buffered LDS (1 barrier per k-chunk).
// ---------------------------------------------------------------------------

typedef __attribute__((ext_vector_type(16))) __bf16 v16bf;
typedef __attribute__((ext_vector_type(8)))  float  v8f;

union FragBF {
    v16bf v;
    uint4 q[2];
};

__device__ __forceinline__ unsigned bf16bits(float f) {
    unsigned u = __float_as_uint(f);
    // round-to-nearest-even fp32 -> bf16
    return (u + 0x7fffu + ((u >> 16) & 1u)) >> 16;
}
__device__ __forceinline__ unsigned pk2(float lo, float hi) {
    return (bf16bits(hi) << 16) | bf16bits(lo);
}
__device__ __forceinline__ unsigned long long shflxor64(unsigned long long v, int m) {
    unsigned lo = (unsigned)v, hi = (unsigned)(v >> 32);
    lo = __shfl_xor(lo, m, 32);
    hi = __shfl_xor(hi, m, 32);
    return ((unsigned long long)hi << 32) | lo;
}
// monotonic float -> u32 key (handles negatives)
__device__ __forceinline__ unsigned sortkey(float f) {
    unsigned b = __float_as_uint(f);
    return (b & 0x80000000u) ? ~b : (b | 0x80000000u);
}

// ---------------------------------------------------------------------------
// 1) fp32 -> bf16 conversion of inputs (8 elements per thread)
// ---------------------------------------------------------------------------
__global__ void cvt_inputs_kernel(const float* __restrict__ X, __bf16* __restrict__ Xb) {
    size_t i = ((size_t)blockIdx.x * blockDim.x + threadIdx.x) * 8;
    float4 a = *(const float4*)(X + i);
    float4 b = *(const float4*)(X + i + 4);
    uint4 p;
    p.x = pk2(a.x, a.y); p.y = pk2(a.z, a.w);
    p.z = pk2(b.x, b.y); p.w = pk2(b.z, b.w);
    *(uint4*)(Xb + i) = p;
}

// ---------------------------------------------------------------------------
// 2) codebook fp32 -> bf16 + per-code squared norms. One block per code row.
//    Assumes D == 1024 (256 threads x 4 elems).
// ---------------------------------------------------------------------------
__global__ void cvt_codebook_norms_kernel(const float* __restrict__ CB,
                                          __bf16* __restrict__ CBb,
                                          float* __restrict__ cbNorm, int D) {
    int k = blockIdx.x;
    int t = threadIdx.x;
    const float4* src = (const float4*)(CB + (size_t)k * D);
    float4 v = src[t];
    uint2 p;
    p.x = pk2(v.x, v.y); p.y = pk2(v.z, v.w);
    *(uint2*)(CBb + (size_t)k * D + (size_t)t * 4) = p;
    float s = v.x * v.x + v.y * v.y + v.z * v.z + v.w * v.w;
    __shared__ float red[256];
    red[t] = s;
    __syncthreads();
    for (int o = 128; o > 0; o >>= 1) {
        if (t < o) red[t] += red[t + o];
        __syncthreads();
    }
    if (t == 0) cbNorm[k] = red[0];
}

// ---------------------------------------------------------------------------
// 3) init per-row argmin slots
// ---------------------------------------------------------------------------
__global__ void init_rowmin_kernel(unsigned long long* __restrict__ rowMin, int N) {
    int i = blockIdx.x * blockDim.x + threadIdx.x;
    if (i < N) rowMin[i] = ~0ull;
}

// ---------------------------------------------------------------------------
// 4) Tiled bf16 WMMA GEMM + fused argmin.
//    Block: 256 threads (8 waves as 4x2), tile 128 rows x 128 codes.
//    Wave (wr,wc) owns rows [32wr,32wr+32) x cols [64wc,64wc+64):
//    2x4 = 8 wmma per k=32 chunk with all fragments preloaded.
//    Double-buffered LDS -> one barrier per chunk.
// ---------------------------------------------------------------------------
__global__ void __launch_bounds__(256)
gemm_argmin_kernel(const __bf16* __restrict__ A,   // [N][D] bf16 inputs
                   const __bf16* __restrict__ B,   // [K][D] bf16 codebook
                   const float* __restrict__ cbNorm,
                   unsigned long long* __restrict__ rowMin,
                   int D) {
    // padded stride 40 (80B) -> 16B-aligned rows, conflict-free ds_load_b128
    __shared__ __bf16 sA[2][128][40];
    __shared__ __bf16 sB[2][128][40];

    const int tid     = threadIdx.x;
    const int rowBase = blockIdx.y * 128;
    const int colBase = blockIdx.x * 128;

    // --- staging mapping: 2 threads per row, 16 bf16 (32B) each ---
    const int srow = tid >> 1;
    const int shlf = (tid & 1) * 16;

    const __bf16* gA = A + (size_t)(rowBase + srow) * D + shlf;
    const __bf16* gB = B + (size_t)(colBase + srow) * D + shlf;

    uint4 pa0, pa1, pb0, pb1;
    {
        const uint4* a = (const uint4*)gA;
        const uint4* b = (const uint4*)gB;
        pa0 = a[0]; pa1 = a[1];
        pb0 = b[0]; pb1 = b[1];
    }
    // stage chunk 0 into buffer 0
    *(uint4*)&sA[0][srow][shlf]     = pa0;
    *(uint4*)&sA[0][srow][shlf + 8] = pa1;
    *(uint4*)&sB[0][srow][shlf]     = pb0;
    *(uint4*)&sB[0][srow][shlf + 8] = pb1;

    // --- per-wave fragment mapping (4x2 wave grid) ---
    const int wv   = tid >> 5;        // wave 0..7
    const int wr   = wv >> 1;         // row group 0..3  (32 rows each)
    const int wc   = wv & 1;          // col group 0..1  (64 cols each)
    const int lane = tid & 31;
    const int lrow = lane & 15;
    const int grp  = lane >> 4;
    const int kk   = grp * 8;         // K sub-offset per ISA 16-bit layout

    v8f acc[2][4] = {};               // 2 m-tiles x 4 c-tiles x 8 fp32 VGPRs

    int cur = 0;
    for (int kc = 0; kc < D; kc += 32) {
        __syncthreads();              // buffer `cur` ready; `cur^1` free
        const bool more = (kc + 32 < D);
        if (more) {                   // prefetch next chunk from global (L2)
            const uint4* a = (const uint4*)(gA + kc + 32);
            const uint4* b = (const uint4*)(gB + kc + 32);
            pa0 = a[0]; pa1 = a[1];
            pb0 = b[0]; pb1 = b[1];
        }

        // preload ALL fragments (single dscnt wait), then WMMA back-to-back
        FragBF fa[2];
#pragma unroll
        for (int mi = 0; mi < 2; ++mi) {
            const int ar = 32 * wr + 16 * mi + lrow;
            fa[mi].q[0] = *(const uint4*)&sA[cur][ar][kk];
            fa[mi].q[1] = *(const uint4*)&sA[cur][ar][kk + 16];
        }
        FragBF fb[4];
#pragma unroll
        for (int ci = 0; ci < 4; ++ci) {
            const int bc = 64 * wc + 16 * ci + lrow;
            fb[ci].q[0] = *(const uint4*)&sB[cur][bc][kk];
            fb[ci].q[1] = *(const uint4*)&sB[cur][bc][kk + 16];
        }
#pragma unroll
        for (int mi = 0; mi < 2; ++mi)
#pragma unroll
            for (int ci = 0; ci < 4; ++ci)
                acc[mi][ci] = __builtin_amdgcn_wmma_f32_16x16x32_bf16(
                    false, fa[mi].v, false, fb[ci].v, (short)0, acc[mi][ci],
                    false, false);

        if (more) {                   // stage prefetched chunk into other buffer
            const int nxt = cur ^ 1;
            *(uint4*)&sA[nxt][srow][shlf]     = pa0;
            *(uint4*)&sA[nxt][srow][shlf + 8] = pa1;
            *(uint4*)&sB[nxt][srow][shlf]     = pb0;
            *(uint4*)&sB[nxt][srow][shlf + 8] = pb1;
            cur = nxt;
        }
    }

    // --- epilogue: key = ||e||^2 - 2*dot (||x||^2 constant per row) ---
    float cn[4];
#pragma unroll
    for (int ci = 0; ci < 4; ++ci)
        cn[ci] = cbNorm[colBase + 64 * wc + 16 * ci + lrow];

#pragma unroll
    for (int mi = 0; mi < 2; ++mi) {
#pragma unroll
        for (int r = 0; r < 8; ++r) {
            unsigned long long best = ~0ull;
#pragma unroll
            for (int ci = 0; ci < 4; ++ci) {
                float dist = __builtin_fmaf(-2.0f, acc[mi][ci][r], cn[ci]);
                unsigned long long packed =
                    ((unsigned long long)sortkey(dist) << 32) |
                    (unsigned)(colBase + 64 * wc + 16 * ci + lrow);
                best = packed < best ? packed : best;
            }
            // min across the 16 lanes sharing this row (xor stays in half-wave)
            best = min(best, shflxor64(best, 1));
            best = min(best, shflxor64(best, 2));
            best = min(best, shflxor64(best, 4));
            best = min(best, shflxor64(best, 8));
            if (lrow == 0) {
                int row = rowBase + 32 * wr + 16 * mi + r + 8 * grp;
                atomicMin(&rowMin[row], best);
            }
        }
    }
}

// ---------------------------------------------------------------------------
// 5) gather quantized = codebook[idx]; per-row loss partials; index output.
//    One block per row n; D == 1024 -> 256 threads x float4.
// ---------------------------------------------------------------------------
__global__ void gather_loss_kernel(const float* __restrict__ X,
                                   const float* __restrict__ CB,
                                   const unsigned long long* __restrict__ rowMin,
                                   float* __restrict__ outQ,
                                   float* __restrict__ outIdx,
                                   float* __restrict__ rowLoss, int D) {
    int n = blockIdx.x;
    int t = threadIdx.x;
    int idx = (int)(unsigned)(rowMin[n] & 0xffffffffu);
    const float4* q4 = (const float4*)(CB + (size_t)idx * D);
    const float4* x4 = (const float4*)(X + (size_t)n * D);
    float4* o4 = (float4*)(outQ + (size_t)n * D);
    float4 q = q4[t];
    float4 x = x4[t];
    o4[t] = q;  // straight-through forward value == quantized
    float dx = q.x - x.x, dy = q.y - x.y, dz = q.z - x.z, dw = q.w - x.w;
    float s = dx * dx + dy * dy + dz * dz + dw * dw;
    __shared__ float red[256];
    red[t] = s;
    __syncthreads();
    for (int o = 128; o > 0; o >>= 1) {
        if (t < o) red[t] += red[t + o];
        __syncthreads();
    }
    if (t == 0) {
        rowLoss[n] = red[0];
        outIdx[n] = (float)idx;
    }
}

// ---------------------------------------------------------------------------
// 6) final deterministic loss reduction
// ---------------------------------------------------------------------------
__global__ void final_loss_kernel(const float* __restrict__ rowLoss,
                                  float* __restrict__ outLoss, int N, int D) {
    int t = threadIdx.x;
    float s = 0.f;
    for (int i = t; i < N; i += 256) s += rowLoss[i];
    __shared__ float red[256];
    red[t] = s;
    __syncthreads();
    for (int o = 128; o > 0; o >>= 1) {
        if (t < o) red[t] += red[t + o];
        __syncthreads();
    }
    if (t == 0) {
        float mean = red[0] / ((float)N * (float)D);
        outLoss[0] = mean;          // q_latent_loss
        outLoss[1] = 0.25f * mean;  // COMMITMENT_COST * e_latent_loss
    }
}

// ---------------------------------------------------------------------------
extern "C" void kernel_launch(void* const* d_in, const int* in_sizes, int n_in,
                              void* d_out, int out_size, void* d_ws, size_t ws_size,
                              hipStream_t stream) {
    const float* X  = (const float*)d_in[0];   // [N,D] fp32
    const float* CB = (const float*)d_in[1];   // [K,D] fp32
    const int D = 1024;
    const int N = in_sizes[0] / D;             // 16384
    const int K = in_sizes[1] / D;             // 8192

    // workspace layout
    char* ws = (char*)d_ws;
    float* cbNorm  = (float*)ws;                                    // K fp32
    float* rowLoss = (float*)(ws + (size_t)K * 4);                  // N fp32
    unsigned long long* rowMin =
        (unsigned long long*)(ws + (size_t)K * 4 + (size_t)N * 4);  // N u64
    __bf16* Xb  = (__bf16*)(ws + (size_t)K * 4 + (size_t)N * 12);   // N*D bf16
    __bf16* CBb = Xb + (size_t)N * D;                               // K*D bf16

    float* outQ    = (float*)d_out;             // [N*D]
    float* outLoss = outQ + (size_t)N * D;      // [2]
    float* outIdx  = outLoss + 2;               // [N]

    cvt_inputs_kernel<<<(N * (D / 2048)), 256, 0, stream>>>(X, Xb); // N*D/2048 blocks
    cvt_codebook_norms_kernel<<<K, 256, 0, stream>>>(CB, CBb, cbNorm, D);
    init_rowmin_kernel<<<(N + 255) / 256, 256, 0, stream>>>(rowMin, N);

    dim3 grid(K / 128, N / 128);
    gemm_argmin_kernel<<<grid, 256, 0, stream>>>(Xb, CBb, cbNorm, rowMin, D);

    gather_loss_kernel<<<N, 256, 0, stream>>>(X, CB, rowMin, outQ, outIdx, rowLoss, D);
    final_loss_kernel<<<1, 256, 0, stream>>>(rowLoss, outLoss, N, D);
}
